// FastWeightGluMLPMultihead_23742579212696
// MI455X (gfx1250) — matrix operand order repaired and checked by reading, not compiled, not run on hardware
//
#include <hip/hip_runtime.h>
#include <hip/hip_bf16.h>
#include <math.h>

// CDNA5 / gfx1250: wave32, v_wmma_f32_16x16x32_bf16, fp32 accumulate.
typedef __attribute__((ext_vector_type(16))) __bf16 v16bf;
typedef __attribute__((ext_vector_type(8)))  __bf16 v8bf;
typedef __attribute__((ext_vector_type(8)))  float  v8f;

#define BLK   512          // 16 wave32 waves
#define NWAVE (BLK / 32)
#define P64   72           // bf16 pitch for K=64 rows: 144B (16B aligned, 36-dword bank step)
#define P128  136          // bf16 pitch for K=128 rows: 272B (16B aligned, 68-dword bank step)

// ---------------------------------------------------------------------------
// WMMA gemm: C[M,N] = A[M,K] * B[K,N] (+ d_scale * Dadd), operands bf16 in LDS.
// A  : M x K row-major, pitch ap  (16B-aligned rows)
// Bt : N x K row-major, pitch bp  (B stored transposed -> contiguous in K)
// Each fragment = two 16B ds_load_b128 per lane; no converts in the loop.
// ---------------------------------------------------------------------------
template <typename OutT>
__device__ __forceinline__ void gemm_bf(
    int tid,
    const __bf16* __restrict__ A,  int ap,
    const __bf16* __restrict__ Bt, int bp,
    OutT* __restrict__ C, int c_ld,
    int M, int N, int K,
    const __bf16* __restrict__ Dadd, int d_ld, float d_scale)
{
  const int lane  = tid & 31;
  const int wave  = tid >> 5;
  const int lhalf = lane >> 4;     // half-wave select
  const int lmod  = lane & 15;
  const int tn     = N >> 4;
  const int ntiles = (M >> 4) * tn;
  for (int t = wave; t < ntiles; t += NWAVE) {   // uniform per wave -> EXEC all ones
    const int tm  = (t / tn) << 4;
    const int tnn = (t % tn) << 4;
    const __bf16* arow = A  + (size_t)(tm  + lmod) * ap;
    const __bf16* brow = Bt + (size_t)(tnn + lmod) * bp;
    v8f acc = {};
    for (int kk = 0; kk < K; kk += 32) {
      // A fragment: j=0..7 -> k = lhalf*8 + j ; j=8..15 -> k = 16 + lhalf*8 + (j-8)
      const v8bf a0 = *(const v8bf*)(arow + kk + (lhalf << 3));
      const v8bf a1 = *(const v8bf*)(arow + kk + 16 + (lhalf << 3));
      // B fragment: j=0..7 -> k = lhalf*8 + j ; j=8..15 -> k = 16 + lhalf*8 + (j-8)
      const v8bf b0 = *(const v8bf*)(brow + kk + (lhalf << 3));
      const v8bf b1 = *(const v8bf*)(brow + kk + 16 + (lhalf << 3));
      const v16bf af = __builtin_shufflevector(a0, a1, 0,1,2,3,4,5,6,7,8,9,10,11,12,13,14,15);
      const v16bf bf_ = __builtin_shufflevector(b0, b1, 0,1,2,3,4,5,6,7,8,9,10,11,12,13,14,15);
      acc = __builtin_amdgcn_wmma_f32_16x16x32_bf16(false, af, false, bf_,
                                                    (short)0, acc, false, false);
    }
    const int cm = tm + (lhalf << 3);
    const int cn = tnn + lmod;
#pragma unroll
    for (int r = 0; r < 8; ++r) {
      float vv = acc[r];
      if (Dadd) vv += d_scale * (float)Dadd[(size_t)(cm + r) * d_ld + cn];
      C[(size_t)(cm + r) * c_ld + cn] = (OutT)vv;
    }
  }
}

// Stage fp32 (arbitrary strides) -> bf16 row-major [R x Kd], pitch dp.
__device__ __forceinline__ void stage_f2b(int tid, const float* __restrict__ S, int rs, int cs,
                                          __bf16* __restrict__ D, int dp, int R, int Kd)
{
  const int tot = R * Kd;
  for (int i = tid; i < tot; i += BLK) {
    const int r = i / Kd, k = i - r * Kd;
    D[r * dp + k] = (__bf16)S[(size_t)r * rs + (size_t)k * cs];
  }
}

// Stage bf16 transpose: D[r*dp+k] = S[k*sp+r]  (D is S^T, R x Kd)
__device__ __forceinline__ void stage_b2bT(int tid, const __bf16* __restrict__ S, int sp,
                                           __bf16* __restrict__ D, int dp, int R, int Kd)
{
  const int tot = R * Kd;
  for (int i = tid; i < tot; i += BLK) {
    const int r = i / Kd, k = i - r * Kd;
    D[r * dp + k] = S[k * sp + r];
  }
}

// Quintic Newton-Schulz, natively bf16 (as the reference). Input X in bX0
// (64x128, pitch P128). bT: 128xP64 scratch, bA: 64xP64 staging,
// sA/sB fp32 64x64. Result added into fp32 W (pitch 128).
__device__ __forceinline__ void newton_schulz5_bf(
    int tid, int steps,
    __bf16* __restrict__ bX0, __bf16* __restrict__ bX1,
    __bf16* __restrict__ bT,  __bf16* __restrict__ bA,
    float* __restrict__ sA, float* __restrict__ sB,
    float* __restrict__ red, float* __restrict__ W)
{
  // X /= (||X||_F + 1e-7)
  if (tid == 0) red[0] = 0.f;
  __syncthreads();
  float s = 0.f;
  for (int i = tid; i < 8192; i += BLK) {
    const float x = (float)bX0[((i >> 7) * P128) + (i & 127)];
    s += x * x;
  }
  atomicAdd(&red[0], s);
  __syncthreads();
  const float inv = 1.f / (sqrtf(red[0]) + 1e-7f);
  for (int i = tid; i < 8192; i += BLK) {
    const int ii = ((i >> 7) * P128) + (i & 127);
    bX0[ii] = (__bf16)((float)bX0[ii] * inv);
  }
  __syncthreads();

  __bf16* Xc = bX0; __bf16* Xn = bX1;
  for (int st = 0; st < steps; ++st) {
    // A = X @ X^T : A-op and Bt-op are both X itself (no staging!)
    gemm_bf<float>(tid, Xc, P128, Xc, P128, sA, 64, 64, 64, 128, nullptr, 0, 0.f);
    __syncthreads();
    stage_f2b(tid, sA, 64, 1, bA, P64, 64, 64);    // bA <- A (bf16)
    __syncthreads();
    // A2 = A @ A : A symmetric -> Bt = bA too
    gemm_bf<float>(tid, bA, P64, bA, P64, sB, 64, 64, 64, 64, nullptr, 0, 0.f);
    __syncthreads();
    // B = b*A + c*A2 (fp32), then stage bA <- B and bT <- X^T
    for (int i = tid; i < 4096; i += BLK) sB[i] = -4.775f * sA[i] + 2.0315f * sB[i];
    __syncthreads();
    stage_f2b(tid, sB, 64, 1, bA, P64, 64, 64);
    stage_b2bT(tid, Xc, P128, bT, P64, 128, 64);   // bT[n,k] = X[k,n]
    __syncthreads();
    // Xn = a*X + B @ X
    gemm_bf<__bf16>(tid, bA, P64, bT, P64, Xn, P128, 64, 128, 64, Xc, P128, 3.4445f);
    __syncthreads();
    __bf16* tp = Xc; Xc = Xn; Xn = tp;
  }
  for (int i = tid; i < 8192; i += BLK)
    W[i] += (float)Xc[((i >> 7) * P128) + (i & 127)];
  __syncthreads();
}

__global__ __launch_bounds__(BLK, 1)
void fw_glu_mlp_kernel(
    const float* __restrict__ q,  const float* __restrict__ k,  const float* __restrict__ v,
    const float* __restrict__ w0, const float* __restrict__ w1, const float* __restrict__ w2,
    const float* __restrict__ lr0, const float* __restrict__ lr1, const float* __restrict__ lr2,
    float* __restrict__ out, float* __restrict__ w0f, float* __restrict__ w1f, float* __restrict__ w2f,
    int L, const int* __restrict__ steps_p)
{
  extern __shared__ float sm[];
  // ---- fp32 region (weight masters + small NS matrices + scalars) ----
  float* sW0  = sm;                  // 64 x 128 (d x dh)
  float* sW1T = sW0  + 8192;         // 64 x 128 (w1^T)
  float* sW2  = sW1T + 8192;         // 64 x 128
  float* sA   = sW2  + 8192;         // 64 x 64  (NS A)
  float* sB   = sA   + 4096;         // 64 x 64  (NS B)
  float* sLr0 = sB   + 4096;         // 64
  float* sLr1 = sLr0 + 64;
  float* sLr2 = sLr1 + 64;
  float* sN0  = sLr2 + 64;           // 128
  float* sN1  = sN0  + 128;          // 64
  float* sN2  = sN1  + 64;           // 128
  float* sRed = sN2  + 128;          // 8
  // ---- bf16 region (all GEMM operands; 16B-aligned buffers) ----
  __bf16* bBase = (__bf16*)(sRed + 8);
  __bf16* bK   = bBase;              // 64 x P64   K row-major
  __bf16* bKT  = bK   + 64 * P64;    // 64 x P64   K^T
  __bf16* bVQ  = bKT  + 64 * P64;    // 64 x P64   V (then Q at output phase)
  __bf16* bVT  = bVQ  + 64 * P64;    // 64 x P64   V^T
  __bf16* bW0T = bVT  + 64 * P64;    // 128 x P64  W0^T mirror
  __bf16* bW2T = bW0T + 128 * P64;   // 128 x P64  W2^T mirror
  __bf16* bG   = bW2T + 128 * P64;   // 64 x P128  gate -> dgate_pre*lr0 -> gate2 -> GH
  __bf16* bH   = bG   + 64 * P128;   // 64 x P128  hpre -> dh_pre_mul*lr2 -> hpre2
  __bf16* bD   = bH   + 64 * P128;   // 64 x P128  dhidden -> hidden*lr1
  __bf16* bX0  = bD   + 64 * P128;   // 64 x P128  NS ping
  __bf16* bX1  = bX0  + 64 * P128;   // 64 x P128  NS pong
  __bf16* bT   = bX1  + 64 * P128;   // 128 x P64 / 64 x P128 shared transpose scratch
  __bf16* bA   = bT   + 128 * P64;   // 64 x P64   NS A/B staging

  const int h     = blockIdx.x;
  const int tid   = threadIdx.x;
  const int steps = steps_p[0];
  const int n_chunks = L >> 6;       // C = 64

  const float* gq  = q  + (size_t)h * L * 64;
  const float* gk  = k  + (size_t)h * L * 64;
  const float* gv  = v  + (size_t)h * L * 64;
  const float* gl0 = lr0 + (size_t)h * L;
  const float* gl1 = lr1 + (size_t)h * L;
  const float* gl2 = lr2 + (size_t)h * L;

  // ---- load initial weights (w1 transposed), build bf16 mirrors ----
  for (int i = tid; i < 8192; i += BLK) {
    sW0[i] = w0[(size_t)h * 8192 + i];
    sW2[i] = w2[(size_t)h * 8192 + i];
    const int r = i >> 6, c = i & 63;          // w1 is [dh=128, d=64] row-major
    sW1T[c * 128 + r] = w1[(size_t)h * 8192 + i];
  }
  __syncthreads();
  for (int i = tid; i < 8192; i += BLK) {
    const int n = i >> 6, kd = i & 63;         // n over dh, kd over d
    bW0T[n * P64 + kd] = (__bf16)sW0[kd * 128 + n];
    bW2T[n * P64 + kd] = (__bf16)sW2[kd * 128 + n];
  }
  // ---- weight-norm targets (norm over axis=1 of original weights) ----
  if (tid < 128) {
    float s = 0.f;
    for (int i = 0; i < 64; ++i) { const float x = sW0[i * 128 + tid]; s += x * x; }
    sN0[tid] = sqrtf(s);
  } else if (tid < 256) {
    const int j = tid - 128; float s = 0.f;
    for (int i = 0; i < 64; ++i) { const float x = sW2[i * 128 + j]; s += x * x; }
    sN2[j] = sqrtf(s);
  } else if (tid < 320) {
    const int j = tid - 256; float s = 0.f;
    for (int i = 0; i < 128; ++i) { const float x = sW1T[j * 128 + i]; s += x * x; }
    sN1[j] = sqrtf(s);
  }
  __syncthreads();

  for (int ch = 0; ch < n_chunks; ++ch) {
    const int base = ch << 6;
    // ---- stage chunk inputs straight to bf16 (both orientations) ----
    for (int i = tid; i < 4096; i += BLK) {
      const int r = i >> 6, c = i & 63;
      const float kv = gk[(size_t)base * 64 + i];
      const float vv = gv[(size_t)base * 64 + i];
      bK [r * P64 + c] = (__bf16)kv;
      bKT[c * P64 + r] = (__bf16)kv;
      bVQ[r * P64 + c] = (__bf16)vv;
      bVT[c * P64 + r] = (__bf16)vv;
    }
    if (tid < 64) {
      sLr0[tid] = gl0[base + tid];
      sLr1[tid] = gl1[base + tid];
      sLr2[tid] = gl2[base + tid];
    }
    // bT <- W1 (Bt for dhidden gemm): bT[n,k] = W1[n,k]^T-source = sW1T[k*128+n]
    stage_f2b(tid, sW1T, 1, 128, bT, P64, 128, 64);
    // prefetch next chunk while this one computes
    if (ch + 1 < n_chunks && tid < 256) {
      const size_t nb = (size_t)(base + 64) * 64;
      __builtin_prefetch(gk + nb + tid * 16, 0, 1);
      __builtin_prefetch(gv + nb + tid * 16, 0, 1);
      __builtin_prefetch(gq + nb + tid * 16, 0, 1);
    }
    __syncthreads();

    // ---- forward/grad GEMMs: gate = K@W0, hpre = K@W2, dhidden = V@W1^T ----
    gemm_bf<__bf16>(tid, bK,  P64, bW0T, P64, bG, P128, 64, 128, 64, nullptr, 0, 0.f);
    gemm_bf<__bf16>(tid, bK,  P64, bW2T, P64, bH, P128, 64, 128, 64, nullptr, 0, 0.f);
    gemm_bf<__bf16>(tid, bVQ, P64, bT,   P64, bD, P128, 64, 128, 64, nullptr, 0, 0.f);
    __syncthreads();

    // ---- elementwise backward (in place, lr folded in) ----
    for (int i = tid; i < 8192; i += BLK) {
      const int row = i >> 7;
      const int ii  = (row * P128) + (i & 127);
      const float g  = (float)bG[ii];
      const float hp = (float)bH[ii];
      const float dh = (float)bD[ii];
      const float sgm = 1.f / (1.f + __expf(-g));
      const float sg  = g * sgm;
      const float dgate_pre = (dh * hp) * sgm * (1.f + g * (1.f - sgm));
      bG[ii] = (__bf16)(dgate_pre * sLr0[row]);   // w0 grad input
      bH[ii] = (__bf16)(dh * sg   * sLr2[row]);   // w2 grad input
      bD[ii] = (__bf16)(sg * hp   * sLr1[row]);   // w1 grad input (hidden*lr1)
    }
    __syncthreads();

    // ---- w0: X = K^T @ G' ; NS5 ; w0 += X ----
    stage_b2bT(tid, bG, P128, bT, P64, 128, 64);            // bT = G'^T
    __syncthreads();
    gemm_bf<__bf16>(tid, bKT, P64, bT, P64, bX0, P128, 64, 128, 64, nullptr, 0, 0.f);
    __syncthreads();
    newton_schulz5_bf(tid, steps, bX0, bX1, bT, bA, sA, sB, sRed, sW0);
    // ---- w1: X = V^T @ D' (= w1g^T) ; NS5 ; w1T += X ----
    stage_b2bT(tid, bD, P128, bT, P64, 128, 64);
    __syncthreads();
    gemm_bf<__bf16>(tid, bVT, P64, bT, P64, bX0, P128, 64, 128, 64, nullptr, 0, 0.f);
    __syncthreads();
    newton_schulz5_bf(tid, steps, bX0, bX1, bT, bA, sA, sB, sRed, sW1T);
    // ---- w2: X = K^T @ H' ; NS5 ; w2 += X ----
    stage_b2bT(tid, bH, P128, bT, P64, 128, 64);
    __syncthreads();
    gemm_bf<__bf16>(tid, bKT, P64, bT, P64, bX0, P128, 64, 128, 64, nullptr, 0, 0.f);
    __syncthreads();
    newton_schulz5_bf(tid, steps, bX0, bX1, bT, bA, sA, sB, sRed, sW2);

    // ---- re-apply weight norm; refresh bf16 mirrors in the same pass ----
    if (tid < 128) {
      const int j = tid; float s = 0.f;
      for (int i = 0; i < 64; ++i) { const float x = sW0[i * 128 + j]; s += x * x; }
      const float sc = sN0[j] / (sqrtf(s) + 1e-5f);
      for (int i = 0; i < 64; ++i) {
        const float x = sW0[i * 128 + j] * sc;
        sW0[i * 128 + j] = x;
        bW0T[j * P64 + i] = (__bf16)x;
      }
    } else if (tid < 256) {
      const int j = tid - 128; float s = 0.f;
      for (int i = 0; i < 64; ++i) { const float x = sW2[i * 128 + j]; s += x * x; }
      const float sc = sN2[j] / (sqrtf(s) + 1e-5f);
      for (int i = 0; i < 64; ++i) {
        const float x = sW2[i * 128 + j] * sc;
        sW2[i * 128 + j] = x;
        bW2T[j * P64 + i] = (__bf16)x;
      }
    } else if (tid < 320) {
      const int j = tid - 256; float s = 0.f;
      for (int i = 0; i < 128; ++i) { const float x = sW1T[j * 128 + i]; s += x * x; }
      const float sc = sN1[j] / (sqrtf(s) + 1e-5f);
      for (int i = 0; i < 128; ++i) sW1T[j * 128 + i] *= sc;
    }
    __syncthreads();

    // ---- output: (silu(Q@W0) * (Q@W2)) @ W1 ----
    for (int i = tid; i < 4096; i += BLK) {       // Q -> bVQ (V no longer needed)
      const int r = i >> 6, c = i & 63;
      bVQ[r * P64 + c] = (__bf16)gq[(size_t)base * 64 + i];
    }
    stage_f2b(tid, sW1T, 128, 1, bT, P128, 64, 128);   // bT = W1T (Bt for out gemm)
    __syncthreads();
    gemm_bf<__bf16>(tid, bVQ, P64, bW0T, P64, bG, P128, 64, 128, 64, nullptr, 0, 0.f);
    gemm_bf<__bf16>(tid, bVQ, P64, bW2T, P64, bH, P128, 64, 128, 64, nullptr, 0, 0.f);
    __syncthreads();
    for (int i = tid; i < 8192; i += BLK) {
      const int ii = ((i >> 7) * P128) + (i & 127);
      const float g = (float)bG[ii];
      const float sgm = 1.f / (1.f + __expf(-g));
      bG[ii] = (__bf16)(g * sgm * (float)bH[ii]);
    }
    __syncthreads();
    gemm_bf<float>(tid, bG, P128, bT, P128,
                   out + (size_t)h * L * 64 + (size_t)base * 64, 64,
                   64, 64, 128, nullptr, 0, 0.f);
    __syncthreads();
  }

  // ---- write back final weights (w1 de-transposed) ----
  for (int i = tid; i < 8192; i += BLK) {
    w0f[(size_t)h * 8192 + i] = sW0[i];
    w2f[(size_t)h * 8192 + i] = sW2[i];
    const int r = i >> 6, c = i & 63;            // w1 element (r in dh, c in d)
    w1f[(size_t)h * 8192 + i] = sW1T[c * 128 + r];
  }
}

extern "C" void kernel_launch(void* const* d_in, const int* in_sizes, int n_in,
                              void* d_out, int out_size, void* d_ws, size_t ws_size,
                              hipStream_t stream) {
  const float* q   = (const float*)d_in[0];
  const float* k   = (const float*)d_in[1];
  const float* v   = (const float*)d_in[2];
  const float* w0  = (const float*)d_in[3];
  const float* w1  = (const float*)d_in[4];
  const float* w2  = (const float*)d_in[5];
  const float* lr0 = (const float*)d_in[6];
  const float* lr1 = (const float*)d_in[7];
  const float* lr2 = (const float*)d_in[8];
  const int* steps_p = (const int*)d_in[10];   // muon_update_steps (device scalar)

  const int BH = in_sizes[3] / (64 * 128);     // w0: [BH, 64, 128]
  const int L  = in_sizes[0] / (BH * 64);      // q:  [BH, L, 64]

  float* o   = (float*)d_out;                  // output [BH, L, 64]
  float* w0f = o   + (size_t)BH * L * 64;
  float* w1f = w0f + (size_t)BH * 8192;
  float* w2f = w1f + (size_t)BH * 8192;

  // fp32: 3*8192 + 2*4096 + 3*64 + 128+64+128 + 8 = 33288 floats
  // bf16: 5*64*P64 + 2*128*P64 + 5*64*P128 + 128*P64 = 94208 elements
  const size_t f32_count  = 3 * 8192 + 2 * 4096 + 3 * 64 + 320 + 8;
  const size_t bf16_count = (size_t)4 * 64 * P64 + 2 * 128 * P64 +
                            (size_t)5 * 64 * P128 + 128 * P64 + 64 * P64;
  const size_t smem = f32_count * 4 + bf16_count * 2;   // = 321568 bytes
  hipFuncSetAttribute(reinterpret_cast<const void*>(fw_glu_mlp_kernel),
                      hipFuncAttributeMaxDynamicSharedMemorySize, (int)smem);
  fw_glu_mlp_kernel<<<dim3(BH), dim3(BLK), smem, stream>>>(
      q, k, v, w0, w1, w2, lr0, lr1, lr2, o, w0f, w1f, w2f, L, steps_p);
}
